// CrossAttention_2388001816997
// MI455X (gfx1250) — matrix operand order repaired and checked by reading, not compiled, not run on hardware
//
#include <hip/hip_runtime.h>

// ---------------------------------------------------------------------------
// CDNA5 (gfx1250) cross-attention: bf16 WMMA + async LDS double buffering.
// ---------------------------------------------------------------------------

typedef __attribute__((ext_vector_type(16))) __bf16 v16bf;
typedef __attribute__((ext_vector_type(8)))  __bf16 bf16x8;
typedef __attribute__((ext_vector_type(8)))  float  v8f;
typedef __attribute__((ext_vector_type(8)))  unsigned char u8x8;

// exact pointee types the async-LDS builtins expect (per hipcc diagnostic)
typedef int v4i_ __attribute__((vector_size(16)));
typedef int v2i_ __attribute__((vector_size(8)));
typedef __attribute__((address_space(1))) v4i_* gp128_t;
typedef __attribute__((address_space(3))) v4i_* lp128_t;
typedef __attribute__((address_space(1))) v2i_* gp64_t;
typedef __attribute__((address_space(3))) v2i_* lp64_t;

#define DEVINL __device__ __forceinline__

#if __has_builtin(__builtin_amdgcn_global_load_async_to_lds_b128)
#define HAS_ASYNC128 1
#endif
#if __has_builtin(__builtin_amdgcn_global_load_async_to_lds_b64)
#define HAS_ASYNC64 1
#endif

DEVINL void cp_async_b128(void* l, const void* g) {
#ifdef HAS_ASYNC128
    __builtin_amdgcn_global_load_async_to_lds_b128((gp128_t)g, (lp128_t)l, 0, 0);
#else
    *(int4*)l = *(const int4*)g;
#endif
}

DEVINL void cp_async_b64(void* l, const void* g) {
#ifdef HAS_ASYNC64
    __builtin_amdgcn_global_load_async_to_lds_b64((gp64_t)g, (lp64_t)l, 0, 0);
#else
    *(long long*)l = *(const long long*)g;
#endif
}

DEVINL void wait_async0() {
#if __has_builtin(__builtin_amdgcn_s_wait_asynccnt)
    __builtin_amdgcn_s_wait_asynccnt(0);
#elif defined(HAS_ASYNC128) || defined(HAS_ASYNC64)
    asm volatile("s_wait_asynccnt 0" ::: "memory");
#endif
}

DEVINL v8f wmma_bf16(v16bf a, v16bf b, v8f c) {
    return __builtin_amdgcn_wmma_f32_16x16x32_bf16(
        false, a, false, b, (short)0, c, false, false);
}

// A-fragment (16x32 bf16) from row-major LDS tile; ISA 7.12.2 16-bit A layout.
DEVINL v16bf lds_a_frag(const __bf16* base, int row, int stride, int kb, int lane) {
    int ko = (lane & 16) ? 8 : 0;
    const __bf16* p = base + row * stride + kb + ko;
    v16bf f;
    ((bf16x8*)&f)[0] = *(const bf16x8*)(p);
    ((bf16x8*)&f)[1] = *(const bf16x8*)(p + 16);
    return f;
}

// B-fragment (32x16 bf16) from Bt row-major LDS tile.
DEVINL v16bf lds_b_frag(const __bf16* base, int n, int stride, int kb, int lane) {
    int ko = (lane & 16) ? 8 : 0;
    const __bf16* p = base + n * stride + kb + ko;
    v16bf f;
    ((bf16x8*)&f)[0] = *(const bf16x8*)(p);
    ((bf16x8*)&f)[1] = *(const bf16x8*)(p + 16);
    return f;
}

// ---------------------------------------------------------------------------
// Elementwise f32 -> bf16
// ---------------------------------------------------------------------------
__global__ void k_convert_bf16(const float* __restrict__ in,
                               __bf16* __restrict__ out, int n) {
    int i = blockIdx.x * blockDim.x + threadIdx.x;
    int i4 = i * 4;
    if (i4 + 3 < n) {
        float4 v = *(const float4*)(in + i4);
        out[i4 + 0] = (__bf16)v.x;
        out[i4 + 1] = (__bf16)v.y;
        out[i4 + 2] = (__bf16)v.z;
        out[i4 + 3] = (__bf16)v.w;
    }
}

// W [K][N] f32 -> Wt [N][K] bf16
__global__ void k_transpose_bf16(const float* __restrict__ W,
                                 __bf16* __restrict__ Wt, int K, int N) {
    int idx = blockIdx.x * blockDim.x + threadIdx.x;
    if (idx < K * N) {
        int n = idx / K;
        int k = idx - n * K;
        Wt[(size_t)n * K + k] = (__bf16)W[(size_t)k * N + n];
    }
}

// mask [B][N][M] u8 -> maskT [B][M][N] u8 (write-coalesced)
__global__ void k_transpose_mask(const unsigned char* __restrict__ in,
                                 unsigned char* __restrict__ out,
                                 int N, int M, long long total) {
    long long idx = (long long)blockIdx.x * blockDim.x + threadIdx.x;
    if (idx < total) {
        int n = (int)(idx % N);
        long long t = idx / N;
        int m = (int)(t % M);
        int b = (int)(t / M);
        out[idx] = in[((long long)b * N + n) * M + m];
    }
}

// ---------------------------------------------------------------------------
// GEMM: C[M][N] = A[M][K] (bf16 row-major) * Bt[N][K] (bf16 row-major Bᵀ)
// 128x128 block tile, 256 threads (8 waves), wave tile 64x32, K-step 32.
// Double-buffered LDS; async global->LDS staging when available.
// ---------------------------------------------------------------------------
#define LDSTRIDE 40  // 32 + 8 halves: 16B-aligned rows, staggered banks

template <bool OUT_BF16>
__global__ __launch_bounds__(256)
void k_gemm_tn(const __bf16* __restrict__ A, const __bf16* __restrict__ Bt,
               int M, int N, int K,
               __bf16* __restrict__ Cb, float* __restrict__ Cf,
               const float* __restrict__ bias) {
    __shared__ __bf16 As[2][128 * LDSTRIDE];
    __shared__ __bf16 Bs[2][128 * LDSTRIDE];

    const int tid  = threadIdx.x;
    const int lane = tid & 31;
    const int wave = tid >> 5;
    const int wm   = wave >> 2;
    const int wn   = wave & 3;
    const int mBase = blockIdx.y * 128;
    const int nBase = blockIdx.x * 128;

    auto stage = [&](int buf, int k0) {
#pragma unroll
        for (int i = 0; i < 2; ++i) {
            int slot = tid + 256 * i;      // 0..511
            int row  = slot >> 2;          // 0..127
            int ch   = (slot & 3) * 8;     // 0,8,16,24
            cp_async_b128(&As[buf][row * LDSTRIDE + ch],
                          &A[(size_t)(mBase + row) * K + k0 + ch]);
            cp_async_b128(&Bs[buf][row * LDSTRIDE + ch],
                          &Bt[(size_t)(nBase + row) * K + k0 + ch]);
        }
    };

    v8f acc[4][2] = {};

    stage(0, 0);
    wait_async0();
    __syncthreads();

    const int nsteps = K >> 5;
    for (int s = 0; s < nsteps; ++s) {
        int cur = s & 1, nxt = cur ^ 1;
        if (s + 1 < nsteps) stage(nxt, (s + 1) << 5);

        v16bf af[4], bfr[2];
#pragma unroll
        for (int jm = 0; jm < 4; ++jm)
            af[jm] = lds_a_frag(As[cur], wm * 64 + jm * 16 + (lane & 15), LDSTRIDE, 0, lane);
#pragma unroll
        for (int jn = 0; jn < 2; ++jn)
            bfr[jn] = lds_b_frag(Bs[cur], wn * 32 + jn * 16 + (lane & 15), LDSTRIDE, 0, lane);
#pragma unroll
        for (int jm = 0; jm < 4; ++jm)
#pragma unroll
            for (int jn = 0; jn < 2; ++jn)
                acc[jm][jn] = wmma_bf16(af[jm], bfr[jn], acc[jm][jn]);

        wait_async0();
        __syncthreads();
    }

    const int rowHalf = (lane & 16) ? 8 : 0;
    const int colLane = lane & 15;
#pragma unroll
    for (int jm = 0; jm < 4; ++jm)
#pragma unroll
        for (int jn = 0; jn < 2; ++jn)
#pragma unroll
            for (int r = 0; r < 8; ++r) {
                int grow = mBase + wm * 64 + jm * 16 + rowHalf + r;
                int gcol = nBase + wn * 32 + jn * 16 + colLane;
                float v = acc[jm][jn][r];
                if (OUT_BF16)
                    Cb[(size_t)grow * N + gcol] = (__bf16)v;
                else
                    Cf[(size_t)grow * N + gcol] = v + bias[gcol];
            }
}

// ---------------------------------------------------------------------------
// Flash attention with softmax1 (implicit extra zero logit).
//   Q  : [B*N, C] bf16, KV : [B*M, 2C] bf16, maskT : [B][M][N] u8
//   Out: [B*N, C] bf16
// Grid (N/64, B*H); 128 threads = 4 waves, each wave owns 16 Q rows.
// Double-buffered K/V/mask chunks with async staging.
// ---------------------------------------------------------------------------
__global__ __launch_bounds__(128)
void k_flash_attn(const __bf16* __restrict__ Q, const __bf16* __restrict__ KV,
                  const unsigned char* __restrict__ maskT,
                  __bf16* __restrict__ Out) {
    constexpr int Nq = 1024, Mk = 4096, C = 1024, D = 64;
    constexpr int FST = 72;   // bf16 row stride (64 + 8)
    constexpr int MST = 72;   // mask byte row stride (8B-aligned for b64)
    __shared__ __bf16 Qs[64 * FST];
    __shared__ __bf16 Ks[2][64 * FST];
    __shared__ __bf16 Vts[2][64 * FST];            // V transposed: [dd][m]
    __shared__ unsigned char Mts[2][64 * MST];     // mask tile: [m][n]
    __shared__ __bf16 Ps[4 * 16 * FST];            // per-wave P tile

    const int tid  = threadIdx.x;
    const int lane = tid & 31;
    const int wave = tid >> 5;
    const int n0   = blockIdx.x * 64;
    const int bh   = blockIdx.y;
    const int b    = bh >> 4;
    const int h    = bh & 15;
    const int rowHalf = (lane & 16) ? 8 : 0;
    const int colLane = lane & 15;
    const float scale = 0.125f;  // d^-0.5, d=64

    auto stage_chunk = [&](int buf, int m0) {
#pragma unroll
        for (int i = 0; i < 4; ++i) {
            int slot = tid + 128 * i;      // 0..511
            int row  = slot >> 3;          // 0..63
            int ch   = (slot & 7) * 8;     // 0..56
            const __bf16* kvrow = KV + (size_t)(b * Mk + m0 + row) * (2 * C);
            cp_async_b128(&Ks[buf][row * FST + ch], &kvrow[h * D + ch]);
            bf16x8 vv = *(const bf16x8*)&kvrow[C + h * D + ch];
#pragma unroll
            for (int e = 0; e < 8; ++e)
                Vts[buf][(ch + e) * FST + row] = vv[e];
        }
        // mask tile: rows = m index (64), cols = n index (64), 8B chunks
#pragma unroll
        for (int i = 0; i < 4; ++i) {
            int slot = tid + 128 * i;      // 0..511
            int row  = slot >> 3;          // 0..63
            int ch   = (slot & 7) * 8;     // 0..56
            cp_async_b64(&Mts[buf][row * MST + ch],
                         &maskT[((size_t)b * Mk + m0 + row) * Nq + n0 + ch]);
        }
    };

    // stage Q tile (straight copy) + chunk 0
#pragma unroll
    for (int i = 0; i < 4; ++i) {
        int slot = tid + 128 * i;
        int row  = slot >> 3;
        int ch   = (slot & 7) * 8;
        cp_async_b128(&Qs[row * FST + ch],
                      &Q[(size_t)(b * Nq + n0 + row) * C + h * D + ch]);
    }
    stage_chunk(0, 0);
    wait_async0();
    __syncthreads();

    v16bf aq[2];
#pragma unroll
    for (int kk = 0; kk < 2; ++kk)
        aq[kk] = lds_a_frag(Qs, wave * 16 + (lane & 15), FST, kk * 32, lane);

    // online-softmax state; softmax1 pad logit pre-seeded: m=0, l=exp(0-0)=1
    float mrun[8], lrun[8];
    v8f o[4] = {};
#pragma unroll
    for (int r = 0; r < 8; ++r) { mrun[r] = 0.0f; lrun[r] = 1.0f; }

    for (int mc = 0; mc < Mk / 64; ++mc) {
        const int cur = mc & 1, nxt = cur ^ 1;
        if (mc + 1 < Mk / 64) stage_chunk(nxt, (mc + 1) * 64);

        // S = Q * K^T (wave's 16 rows x 64 cols)
        v8f s[4] = {};
#pragma unroll
        for (int kk = 0; kk < 2; ++kk)
#pragma unroll
            for (int j = 0; j < 4; ++j) {
                v16bf kf = lds_b_frag(Ks[cur], j * 16 + colLane, FST, kk * 32, lane);
                s[j] = wmma_bf16(aq[kk], kf, s[j]);
            }

        // scale + mask: one ds_load_b64 per j covers this lane's 8 rows
#pragma unroll
        for (int j = 0; j < 4; ++j) {
            u8x8 mb = *(const u8x8*)&Mts[cur][(j * 16 + colLane) * MST +
                                             wave * 16 + rowHalf];
#pragma unroll
            for (int r = 0; r < 8; ++r) {
                float v = s[j][r] * scale;
                if (mb[r]) v = -1e30f;
                s[j][r] = v;
            }
        }

        // online softmax (row spread across 16 lanes of the half-wave)
#pragma unroll
        for (int r = 0; r < 8; ++r) {
            float mx = -1e30f;
#pragma unroll
            for (int j = 0; j < 4; ++j) mx = fmaxf(mx, s[j][r]);
#pragma unroll
            for (int off = 1; off < 16; off <<= 1)
                mx = fmaxf(mx, __shfl_xor(mx, off, 16));
            float mnew = fmaxf(mrun[r], mx);
            float corr = __expf(mrun[r] - mnew);
            mrun[r] = mnew;
            float rs = 0.0f;
#pragma unroll
            for (int j = 0; j < 4; ++j) {
                float p = __expf(s[j][r] - mnew);
                s[j][r] = p;
                rs += p;
            }
#pragma unroll
            for (int off = 1; off < 16; off <<= 1)
                rs += __shfl_xor(rs, off, 16);
            lrun[r] = lrun[r] * corr + rs;
#pragma unroll
            for (int j2 = 0; j2 < 4; ++j2) o[j2][r] *= corr;
        }

        // P (C/D layout) -> per-wave LDS tile -> A-layout fragments
        __bf16* Pw = Ps + wave * 16 * FST;
#pragma unroll
        for (int j = 0; j < 4; ++j)
#pragma unroll
            for (int r = 0; r < 8; ++r)
                Pw[(rowHalf + r) * FST + j * 16 + colLane] = (__bf16)s[j][r];

        // O += P * V
#pragma unroll
        for (int kk = 0; kk < 2; ++kk) {
            v16bf pf = lds_a_frag(Pw, lane & 15, FST, kk * 32, lane);
#pragma unroll
            for (int j2 = 0; j2 < 4; ++j2) {
                v16bf vf = lds_b_frag(Vts[cur], j2 * 16 + colLane, FST, kk * 32, lane);
                o[j2] = wmma_bf16(pf, vf, o[j2]);
            }
        }

        wait_async0();
        __syncthreads();
    }

    // normalize and write [B*N, C] bf16
#pragma unroll
    for (int r = 0; r < 8; ++r) {
        float inv = 1.0f / lrun[r];
        int gn = n0 + wave * 16 + rowHalf + r;
#pragma unroll
        for (int j2 = 0; j2 < 4; ++j2)
            Out[(size_t)(b * Nq + gn) * C + h * D + j2 * 16 + colLane] =
                (__bf16)(o[j2][r] * inv);
    }
}

// ---------------------------------------------------------------------------
// Launch
// ---------------------------------------------------------------------------
extern "C" void kernel_launch(void* const* d_in, const int* in_sizes, int n_in,
                              void* d_out, int out_size, void* d_ws, size_t ws_size,
                              hipStream_t stream) {
    (void)in_sizes; (void)n_in; (void)out_size; (void)ws_size;
    constexpr int B = 2, N = 1024, M = 4096, C = 1024;

    const float* x     = (const float*)d_in[0];
    const float* ctx   = (const float*)d_in[1];
    const unsigned char* mask = (const unsigned char*)d_in[2];
    const float* Wq    = (const float*)d_in[3];
    const float* Wkv   = (const float*)d_in[4];
    const float* Wproj = (const float*)d_in[5];
    const float* bproj = (const float*)d_in[6];
    float* out = (float*)d_out;

    char* ws = (char*)d_ws;
    size_t off = 0;
    auto alloc = [&](size_t bytes) { char* p = ws + off; off += (bytes + 255) & ~size_t(255); return p; };
    __bf16* xb   = (__bf16*)alloc((size_t)B * N * C * 2);
    __bf16* cb   = (__bf16*)alloc((size_t)B * M * C * 2);
    __bf16* wqt  = (__bf16*)alloc((size_t)C * C * 2);
    __bf16* wkvt = (__bf16*)alloc((size_t)C * 2 * C * 2);
    __bf16* wpt  = (__bf16*)alloc((size_t)C * C * 2);
    __bf16* Qb   = (__bf16*)alloc((size_t)B * N * C * 2);
    __bf16* KVb  = (__bf16*)alloc((size_t)B * M * 2 * C * 2);
    __bf16* AOb  = (__bf16*)alloc((size_t)B * N * C * 2);
    unsigned char* maskT = (unsigned char*)alloc((size_t)B * M * N);

    {
        int n = B * N * C;
        k_convert_bf16<<<(n / 4 + 255) / 256, 256, 0, stream>>>(x, xb, n);
        n = B * M * C;
        k_convert_bf16<<<(n / 4 + 255) / 256, 256, 0, stream>>>(ctx, cb, n);
    }
    k_transpose_bf16<<<(C * C + 255) / 256, 256, 0, stream>>>(Wq, wqt, C, C);
    k_transpose_bf16<<<(C * 2 * C + 255) / 256, 256, 0, stream>>>(Wkv, wkvt, C, 2 * C);
    k_transpose_bf16<<<(C * C + 255) / 256, 256, 0, stream>>>(Wproj, wpt, C, C);
    {
        long long total = (long long)B * M * N;
        k_transpose_mask<<<(unsigned)((total + 255) / 256), 256, 0, stream>>>(
            mask, maskT, N, M, total);
    }

    // Q = x @ Wq : [2048 x 1024]
    k_gemm_tn<true><<<dim3(C / 128, (B * N) / 128), 256, 0, stream>>>(
        xb, wqt, B * N, C, C, Qb, nullptr, nullptr);
    // KV = ctx @ Wkv : [8192 x 2048]
    k_gemm_tn<true><<<dim3((2 * C) / 128, (B * M) / 128), 256, 0, stream>>>(
        cb, wkvt, B * M, 2 * C, C, KVb, nullptr, nullptr);
    // flash attention
    k_flash_attn<<<dim3(N / 64, B * 16), 128, 0, stream>>>(Qb, KVb, maskT, AOb);
    // out = attnOut @ Wproj + bproj : f32
    k_gemm_tn<false><<<dim3(C / 128, (B * N) / 128), 256, 0, stream>>>(
        AOb, wpt, B * N, C, C, nullptr, out, bproj);
}
